// Encoder_18107582120037
// MI455X (gfx1250) — compile-verified, matrix operand order checked by reading, not run on hardware
//
#include <hip/hip_runtime.h>
#include <hip/hip_bf16.h>

// ---------------- problem constants (fixed by setup_inputs) ----------------
constexpr int Sdim  = 4096;   // sequence length
constexpr int Hdim  = 1024;   // hidden size
constexpr int NHEAD = 32;
constexpr int DK    = 32;     // Hdim / NHEAD
constexpr int QA    = 2049;   // split point
constexpr int Bn    = 3;      // batches

// ---------------- WMMA types ----------------
typedef __attribute__((ext_vector_type(16))) __bf16 v16bf;
typedef __attribute__((ext_vector_type(8)))  float  v8f;

union AFrag { v16bf v; unsigned u[8]; };
union CFrag { v8f   v; float    f[8]; };

// A/B 16-bit fragment K-pair offset for VGPR v, half = lane>>4
__device__ __forceinline__ int kpair(int v, int hf) {
    return ((v < 4) ? (2 * v) : (16 + 2 * (v - 4))) + 8 * hf;
}

__device__ __forceinline__ unsigned short f2bf(float x) {
    __bf16 h = (__bf16)x;
    return __builtin_bit_cast(unsigned short, h);
}

// ---------------- workspace layout (bytes) ----------------
constexpr size_t XBF_OFF  = 0;                                          // 3*S*H bf16
constexpr size_t WALL_OFF = XBF_OFF  + (size_t)Bn * Sdim * Hdim * 2;    // 4*H*H bf16 (Wq|Wk|Wv|Wo)
constexpr size_t QKV_OFF  = WALL_OFF + (size_t)4 * Hdim * Hdim * 2;     // 9*S*H bf16 (Q[3b]|K[3b]|V[3b])
constexpr size_t ATT_OFF  = QKV_OFF  + (size_t)9 * Sdim * Hdim * 2;     // 3*S*H bf16
constexpr size_t ROW_OFF  = ATT_OFF  + (size_t)Bn * Sdim * Hdim * 2;    // 3*S*H f32
constexpr size_t WS_NEED  = ROW_OFF  + (size_t)Bn * Sdim * Hdim * 4;    // ~176 MB

// ============================================================
// Kernel 1: f32 -> bf16 conversion of activations and weights
// ============================================================
__global__ void prep_kernel(const float* __restrict__ hs,
                            const float* __restrict__ Wq, const float* __restrict__ Wk,
                            const float* __restrict__ Wv, const float* __restrict__ Wo,
                            unsigned short* __restrict__ Xbf,
                            unsigned short* __restrict__ Wall) {
    const size_t nX = (size_t)Bn * Sdim * Hdim;
    const size_t nW = (size_t)4 * Hdim * Hdim;
    const size_t total = nX + nW;
    for (size_t i = (size_t)blockIdx.x * blockDim.x + threadIdx.x; i < total;
         i += (size_t)gridDim.x * blockDim.x) {
        if (i < nX) {
            Xbf[i] = f2bf(hs[i]);
        } else {
            size_t j = i - nX;
            int w = (int)(j / ((size_t)Hdim * Hdim));
            size_t r = j - (size_t)w * Hdim * Hdim;
            const float* W = (w == 0) ? Wq : (w == 1) ? Wk : (w == 2) ? Wv : Wo;
            Wall[j] = f2bf(W[r]);
        }
    }
}

// ============================================================
// Kernel 2: QKV projection  O = X @ W^T + bias   (bf16 out)
//   grid: (Ntiles/8, Mtiles, 3 batches * 3 weights), block 256 (8 waves)
// ============================================================
__global__ void proj_kernel(const unsigned short* __restrict__ Xbf,
                            const unsigned short* __restrict__ Wall,
                            const float* __restrict__ bq, const float* __restrict__ bk,
                            const float* __restrict__ bv,
                            unsigned short* __restrict__ QKV) {
    const int wave = threadIdx.x >> 5, lane = threadIdx.x & 31;
    const int hf = lane >> 4, c16 = lane & 15;
    const int z = blockIdx.z, b = z / 3, w = z % 3;
    const unsigned short* A = Xbf + (size_t)b * Sdim * Hdim;
    const unsigned short* W = Wall + (size_t)w * Hdim * Hdim;
    const float* bias = (w == 0) ? bq : (w == 1) ? bk : bv;
    unsigned short* O = QKV + ((size_t)w * Bn + b) * (size_t)Sdim * Hdim;

    const int n0 = (blockIdx.x * 8 + wave) * 16;
    const int m0 = blockIdx.y * 16;

    CFrag c;
#pragma unroll
    for (int i = 0; i < 8; ++i) c.f[i] = 0.f;

    for (int kb = 0; kb < Hdim; kb += 32) {
        AFrag a, bf;
#pragma unroll
        for (int v = 0; v < 8; ++v) {
            int k = kpair(v, hf);
            a.u[v]  = *(const unsigned*)(A + (size_t)(m0 + c16) * Hdim + kb + k);
            bf.u[v] = *(const unsigned*)(W + (size_t)(n0 + c16) * Hdim + kb + k);
        }
        c.v = __builtin_amdgcn_wmma_f32_16x16x32_bf16(false, a.v, false, bf.v,
                                                      (short)0, c.v, false, false);
    }
    const float bb = bias[n0 + c16];
#pragma unroll
    for (int v = 0; v < 8; ++v) {
        int row = m0 + v + 8 * hf;
        O[(size_t)row * Hdim + n0 + c16] = f2bf(c.f[v] + bb);
    }
}

// ============================================================
// Kernel 3: flash attention, one wave per (head, 16-query tile)
//   grid: (NHEAD/8, 128 qtiles, 3 batches * 2 directions), block 256
// ============================================================
__global__ void attn_kernel(const unsigned short* __restrict__ Qb_,
                            const unsigned short* __restrict__ Kb_,
                            const unsigned short* __restrict__ Vb_,
                            unsigned short* __restrict__ Ob_) {
    __shared__ unsigned short Pld[8][16 * 32];   // per-wave P tile (16x32 bf16)

    const int wave = threadIdx.x >> 5, lane = threadIdx.x & 31;
    const int hf = lane >> 4, c16 = lane & 15;
    const int head = blockIdx.x * 8 + wave;
    const int qt = blockIdx.y;
    const int z = blockIdx.z, b = z >> 1, which = z & 1;

    const int qStart = which ? QA : 1;
    const int nQ     = which ? (Sdim - QA) : (QA - 1);   // 2047 : 2048
    const int kStart = which ? 1 : QA;
    const int nK     = which ? (QA - 1) : (Sdim - QA);   // 2048 : 2047

    const size_t base = (size_t)b * Sdim * Hdim;
    const unsigned short* Q = Qb_ + base;
    const unsigned short* K = Kb_ + base;
    const unsigned short* V = Vb_ + base;
    unsigned short* O = Ob_ + base;
    const int hc = head * DK;

    const int q0 = qStart + qt * 16;
    int qRow = q0 + c16;
    const int qMax = qStart + nQ - 1;
    if (qRow > qMax) qRow = qMax;

    AFrag qf;
#pragma unroll
    for (int v = 0; v < 8; ++v)
        qf.u[v] = *(const unsigned*)(Q + (size_t)qRow * Hdim + hc + kpair(v, hf));

    float m[8], l[8];
    CFrag acc0, acc1;
#pragma unroll
    for (int v = 0; v < 8; ++v) { m[v] = -3.0e38f; l[v] = 0.f; acc0.f[v] = 0.f; acc1.f[v] = 0.f; }
    const float scale = 0.17677669529663687f;  // 1/sqrt(32)

    for (int kb = 0; kb < nK; kb += 32) {
        // ---- scores: Q(16x32) x K^T(32x32keys) as two 16x16 WMMAs ----
        AFrag kf0, kf1;
        const int kr0 = kb + c16;       const int kr0c = (kr0 < nK) ? kr0 : nK - 1;
        const int kr1 = kb + 16 + c16;  const int kr1c = (kr1 < nK) ? kr1 : nK - 1;
#pragma unroll
        for (int v = 0; v < 8; ++v) {
            int k = kpair(v, hf);
            kf0.u[v] = *(const unsigned*)(K + (size_t)(kStart + kr0c) * Hdim + hc + k);
            kf1.u[v] = *(const unsigned*)(K + (size_t)(kStart + kr1c) * Hdim + hc + k);
        }
        CFrag s0, s1;
#pragma unroll
        for (int v = 0; v < 8; ++v) { s0.f[v] = 0.f; s1.f[v] = 0.f; }
        s0.v = __builtin_amdgcn_wmma_f32_16x16x32_bf16(false, qf.v, false, kf0.v,
                                                       (short)0, s0.v, false, false);
        s1.v = __builtin_amdgcn_wmma_f32_16x16x32_bf16(false, qf.v, false, kf1.v,
                                                       (short)0, s1.v, false, false);
        const bool ko0 = (kr0 >= nK), ko1 = (kr1 >= nK);

        // ---- online softmax over this 32-key block ----
        float t0[8], t1[8];
#pragma unroll
        for (int v = 0; v < 8; ++v) {
            t0[v] = ko0 ? -1.0e30f : s0.f[v] * scale;
            t1[v] = ko1 ? -1.0e30f : s1.f[v] * scale;
        }
#pragma unroll
        for (int v = 0; v < 8; ++v) {
            float bm = fmaxf(t0[v], t1[v]);
            bm = fmaxf(bm, __shfl_xor(bm, 1, 32));
            bm = fmaxf(bm, __shfl_xor(bm, 2, 32));
            bm = fmaxf(bm, __shfl_xor(bm, 4, 32));
            bm = fmaxf(bm, __shfl_xor(bm, 8, 32));
            const float mn = fmaxf(m[v], bm);
            const float al = __expf(m[v] - mn);
            m[v] = mn;
            const float p0 = __expf(t0[v] - mn);
            const float p1 = __expf(t1[v] - mn);
            float ps = p0 + p1;
            ps += __shfl_xor(ps, 1, 32);
            ps += __shfl_xor(ps, 2, 32);
            ps += __shfl_xor(ps, 4, 32);
            ps += __shfl_xor(ps, 8, 32);
            l[v] = l[v] * al + ps;
            acc0.f[v] *= al;
            acc1.f[v] *= al;
            // C-layout -> LDS P tile (row-major 16x32 bf16)
            const int r = v + 8 * hf;
            Pld[wave][r * 32 + c16]      = f2bf(p0);
            Pld[wave][r * 32 + 16 + c16] = f2bf(p1);
        }
        asm volatile("s_wait_dscnt 0" ::: "memory");   // DS write -> DS read, same wave

        // reload P as A-fragment
        AFrag pf;
#pragma unroll
        for (int v = 0; v < 8; ++v)
            pf.u[v] = *(const unsigned*)(&Pld[wave][c16 * 32 + kpair(v, hf)]);

        // ---- PV: P(16x32keys) x V(32keys x 32d) as two WMMAs ----
        AFrag vf0, vf1;
#pragma unroll
        for (int v = 0; v < 8; ++v) {
            const int k = kpair(v, hf);
            int ka = kb + k;     if (ka > nK - 1) ka = nK - 1;
            int kc = kb + k + 1; if (kc > nK - 1) kc = nK - 1;
            const unsigned short* Va = V + (size_t)(kStart + ka) * Hdim + hc;
            const unsigned short* Vc = V + (size_t)(kStart + kc) * Hdim + hc;
            vf0.u[v] = (unsigned)Va[c16]      | ((unsigned)Vc[c16] << 16);
            vf1.u[v] = (unsigned)Va[16 + c16] | ((unsigned)Vc[16 + c16] << 16);
        }
        acc0.v = __builtin_amdgcn_wmma_f32_16x16x32_bf16(false, pf.v, false, vf0.v,
                                                         (short)0, acc0.v, false, false);
        acc1.v = __builtin_amdgcn_wmma_f32_16x16x32_bf16(false, pf.v, false, vf1.v,
                                                         (short)0, acc1.v, false, false);
    }

    // ---- normalize + store (guard tail query rows) ----
#pragma unroll
    for (int v = 0; v < 8; ++v) {
        const int row = q0 + v + 8 * hf;
        if (row < qStart + nQ) {
            const float inv = 1.0f / l[v];
            O[(size_t)row * Hdim + hc + c16]      = f2bf(acc0.f[v] * inv);
            O[(size_t)row * Hdim + hc + 16 + c16] = f2bf(acc1.f[v] * inv);
        }
    }
}

// ============================================================
// Kernel 4: output projection  row = attnOut @ Wo^T + bo  (f32 out)
// ============================================================
__global__ void outproj_kernel(const unsigned short* __restrict__ Abf,
                               const unsigned short* __restrict__ Wobf,
                               const float* __restrict__ bo,
                               float* __restrict__ rowOut) {
    const int wave = threadIdx.x >> 5, lane = threadIdx.x & 31;
    const int hf = lane >> 4, c16 = lane & 15;
    const int b = blockIdx.z;
    const unsigned short* A = Abf + (size_t)b * Sdim * Hdim;
    float* O = rowOut + (size_t)b * Sdim * Hdim;

    const int n0 = (blockIdx.x * 8 + wave) * 16;
    const int m0 = blockIdx.y * 16;

    CFrag c;
#pragma unroll
    for (int i = 0; i < 8; ++i) c.f[i] = 0.f;

    for (int kb = 0; kb < Hdim; kb += 32) {
        AFrag a, bf;
#pragma unroll
        for (int v = 0; v < 8; ++v) {
            int k = kpair(v, hf);
            a.u[v]  = *(const unsigned*)(A + (size_t)(m0 + c16) * Hdim + kb + k);
            bf.u[v] = *(const unsigned*)(Wobf + (size_t)(n0 + c16) * Hdim + kb + k);
        }
        c.v = __builtin_amdgcn_wmma_f32_16x16x32_bf16(false, a.v, false, bf.v,
                                                      (short)0, c.v, false, false);
    }
    const float bb = bo[n0 + c16];
#pragma unroll
    for (int v = 0; v < 8; ++v) {
        int row = m0 + v + 8 * hf;
        O[(size_t)row * Hdim + n0 + c16] = c.f[v] + bb;
    }
}

// ============================================================
// Kernel 5: sentence means -> output (3 x 2048)
//   V11 = mean rows [1,2049), V21 = mean rows [2049,4096)
// ============================================================
__global__ void mean_kernel(const float* __restrict__ rowOut, float* __restrict__ out) {
    const int col = blockIdx.x * blockDim.x + threadIdx.x;
    const int b = blockIdx.y;
    if (col >= Hdim) return;
    const float* R = rowOut + (size_t)b * Sdim * Hdim;
    float s1 = 0.f;
    for (int r = 1; r < QA; ++r) s1 += R[(size_t)r * Hdim + col];
    float s2 = 0.f;
    for (int r = QA; r < Sdim; ++r) s2 += R[(size_t)r * Hdim + col];
    out[(size_t)b * 2048 + col]        = s1 * (1.0f / (float)(QA - 1));
    out[(size_t)b * 2048 + 1024 + col] = s2 * (1.0f / (float)(Sdim - QA));
}

// ============================================================
extern "C" void kernel_launch(void* const* d_in, const int* in_sizes, int n_in,
                              void* d_out, int out_size, void* d_ws, size_t ws_size,
                              hipStream_t stream) {
    // input order: hidden_states, st_mask, edges_src, edges_tgt, edges_type,
    //              edges_pos, all_sen, Wq, bq, Wk, bk, Wv, bv, Wo, bo
    const float* hs = (const float*)d_in[0];
    const float* Wq = (const float*)d_in[7];
    const float* bq = (const float*)d_in[8];
    const float* Wk = (const float*)d_in[9];
    const float* bk = (const float*)d_in[10];
    const float* Wv = (const float*)d_in[11];
    const float* bv = (const float*)d_in[12];
    const float* Wo = (const float*)d_in[13];
    const float* bo = (const float*)d_in[14];
    float* out = (float*)d_out;

    if (ws_size < WS_NEED) return;  // scratch undersized: nothing safe to do

    char* ws = (char*)d_ws;
    unsigned short* Xbf  = (unsigned short*)(ws + XBF_OFF);
    unsigned short* Wall = (unsigned short*)(ws + WALL_OFF);
    unsigned short* QKV  = (unsigned short*)(ws + QKV_OFF);
    unsigned short* Att  = (unsigned short*)(ws + ATT_OFF);
    float*          Row  = (float*)(ws + ROW_OFF);

    const unsigned short* Qbf = QKV;
    const unsigned short* Kbf = QKV + (size_t)Bn * Sdim * Hdim;
    const unsigned short* Vbf = QKV + (size_t)2 * Bn * Sdim * Hdim;

    prep_kernel<<<8192, 256, 0, stream>>>(hs, Wq, Wk, Wv, Wo, Xbf, Wall);

    // QKV projections: N tiles = 1024/16 = 64 -> grid.x = 8 (8 waves/block)
    proj_kernel<<<dim3(8, Sdim / 16, Bn * 3), 256, 0, stream>>>(Xbf, Wall, bq, bk, bv, QKV);

    // attention: 32 heads / 8 waves = 4 blocks.x, 128 query tiles, 3 batches x 2 dirs
    attn_kernel<<<dim3(NHEAD / 8, 128, Bn * 2), 256, 0, stream>>>(Qbf, Kbf, Vbf, Att);

    // output projection with Wo (4th matrix in Wall)
    outproj_kernel<<<dim3(8, Sdim / 16, Bn), 256, 0, stream>>>(
        Att, Wall + (size_t)3 * Hdim * Hdim, bo, Row);

    mean_kernel<<<dim3(Hdim / 256, Bn), 256, 0, stream>>>(Row, out);
}
